// W8A16Linear_87823491269073
// MI455X (gfx1250) — matrix operand order repaired and checked by reading, not compiled, not run on hardware
//
#include <hip/hip_runtime.h>
#include <stdint.h>

// ---------------------------------------------------------------------------
// W8A16 linear: out[m,n] = (sum_k x[m,k] * w8[n,k]) * scale[n] + bias[n]
// x: fp16 [M,K], w8: int8 in int32 containers [N,K], scale/bias fp16.
// M=8192, N=16384, K=4096.  Compute-bound fp16 WMMA GEMM on gfx1250.
//
// Pipeline: pre-pass dequant W -> fp16 workspace (128MB, ~L2-resident), then
// a TRIPLE-buffered LDS GEMM: GLOBAL_LOAD_ASYNC_TO_LDS_B128 staging (ASYNCcnt,
// 2 tiles of prefetch distance via s_wait_asynccnt 8) feeding
// V_WMMA_F32_16X16X32_F16.  4 waves/block, each wave owns a 64x64 output tile
// (16 accumulators) -> 16 WMMA per 16 ds_load_b128 per K-step.
// ---------------------------------------------------------------------------

typedef __attribute__((ext_vector_type(16))) _Float16 v16h;
typedef __attribute__((ext_vector_type(8)))  _Float16 v8h;
typedef __attribute__((ext_vector_type(4)))  _Float16 v4h;
typedef __attribute__((ext_vector_type(8)))  float    v8f;
typedef __attribute__((ext_vector_type(4)))  int      v4i;

#define M_TOT 8192
#define N_TOT 16384
#define K_TOT 4096

constexpr int BM   = 128;
constexpr int BN   = 128;
constexpr int BK   = 32;            // one WMMA K step
constexpr int LDT  = BK + 8;        // padded LDS row stride in halves (80 B)
constexpr int NBUF = 3;             // 3-deep async pipeline
constexpr int KTILES  = K_TOT / BK; // 128
constexpr int NGRID_N = N_TOT / BN; // 128
constexpr int NGRID_M = M_TOT / BM; // 64
constexpr int GROUP_M = 8;          // L2 swizzle: 8 M-tiles share streamed W

// ---- CDNA5 async global->LDS copy (ASYNCcnt), ISA 15.18.3 op 98 -----------
__device__ __forceinline__ void async_ld_b128(uint32_t lds_off, const void* gaddr) {
    asm volatile("global_load_async_to_lds_b128 %0, %1, off"
                 :: "v"(lds_off), "v"(gaddr) : "memory");
}
__device__ __forceinline__ void wait_async8() {   // newest stage may stay in flight
    asm volatile("s_wait_asynccnt 8" ::: "memory");
}
__device__ __forceinline__ void wait_async0() {
    asm volatile("s_wait_asynccnt 0" ::: "memory");
}
__device__ __forceinline__ uint32_t lds_off_of(const void* p) {
    // Generic LDS address: addr[31:0] is the LDS byte offset (ISA 10.2).
    return (uint32_t)(uintptr_t)p;
}

// ---------------------------------------------------------------------------
// Pre-pass: dequantize int32-container int8 weights -> fp16 (bandwidth bound).
// ---------------------------------------------------------------------------
__global__ void __launch_bounds__(256)
dequant_w8_f16(const v4i* __restrict__ w, v4h* __restrict__ wh, int n4) {
    int i = blockIdx.x * 256 + threadIdx.x;
    if (i < n4) {
        v4i v = w[i];
        v4h h;
        h.x = (_Float16)v.x;
        h.y = (_Float16)v.y;
        h.z = (_Float16)v.z;
        h.w = (_Float16)v.w;
        wh[i] = h;
    }
}

// ---------------------------------------------------------------------------
// Tiled WMMA GEMM.  INLINE_DEQ=false: fp16 weights in workspace, async staging.
// INLINE_DEQ=true: int32-container weights converted while staging (fallback).
// ---------------------------------------------------------------------------
template <bool INLINE_DEQ>
__global__ void __launch_bounds__(128)
w8a16_gemm(const _Float16* __restrict__ x,
           const void*     __restrict__ wraw,
           const _Float16* __restrict__ scales,
           const _Float16* __restrict__ bias,
           _Float16*       __restrict__ out) {
    __shared__ _Float16 sA[NBUF][BM * LDT];   // 3 x 10 KB
    __shared__ _Float16 sB[NBUF][BN * LDT];   // 3 x 10 KB  (60 KB total)

    const int tid   = threadIdx.x;        // 128 threads = 4 waves
    const int lane  = tid & 31;
    const int wave  = tid >> 5;
    const int waveM = wave >> 1;          // 0..1 -> 64-row slab
    const int waveN = wave & 1;           // 0..1 -> 64-col slab
    const int ln    = lane & 15;
    const int hf    = lane >> 4;          // lane-half selects K sub-block

    // ---- L2-friendly swizzle: 8 M-tiles per group sweep all N tiles ----
    const int bid   = blockIdx.x;
    const int group = bid / (GROUP_M * NGRID_N);
    const int rem   = bid % (GROUP_M * NGRID_N);
    const int mBase = (group * GROUP_M + (rem % GROUP_M)) * BM;
    const int nBase = (rem / GROUP_M) * BN;

    const _Float16* wh = (const _Float16*)wraw;
    const int*      wi = (const int*)wraw;

    // staging indices: thread covers (t>>2)-th row, 16B chunk (t&3); 4 passes
    const int srow = tid >> 2;            // 0..31
    const int scol = (tid & 3) * 8;       // half offset within BK

    v8f acc[4][4] = {};

    if constexpr (!INLINE_DEQ) {
        // ---- per-thread staging pointers, advanced BK halves per stage ----
        const _Float16* xp = x  + (size_t)(mBase + srow) * K_TOT + scol;
        const _Float16* wp = wh + (size_t)(nBase + srow) * K_TOT + scol;
        const uint32_t aoff0 = lds_off_of(&sA[0][srow * LDT + scol]);
        const uint32_t boff0 = lds_off_of(&sB[0][srow * LDT + scol]);
        constexpr uint32_t BUFBYTES = (uint32_t)(BM * LDT * sizeof(_Float16));
        constexpr uint32_t ROWBYTES = (uint32_t)(32 * LDT * sizeof(_Float16));

        auto stage = [&](int buf) {
            const uint32_t ab = aoff0 + (uint32_t)buf * BUFBYTES;
            const uint32_t bb = boff0 + (uint32_t)buf * BUFBYTES;
#pragma unroll
            for (int i = 0; i < 4; ++i) {
                async_ld_b128(ab + i * ROWBYTES, xp + (size_t)i * 32 * K_TOT);
                async_ld_b128(bb + i * ROWBYTES, wp + (size_t)i * 32 * K_TOT);
            }
            xp += BK;
            wp += BK;
        };

        stage(0);             // k-step 0
        stage(1);             // k-step 1

        for (int kt = 0; kt < KTILES; ++kt) {
            const int cur = kt % NBUF;

            // stage kt must have landed; allow newest stage (8 asyncs) in flight
            if (kt + 1 < KTILES) wait_async8();
            else                 wait_async0();
            __syncthreads();  // all waves: stage kt landed, buf (kt+2)%3 free

            if (kt + 2 < KTILES) stage((kt + 2) % NBUF);

            // ---- fragments in contiguous k order (k = 16*hf + s), same for A and B ----
            v16h afrag[4];
#pragma unroll
            for (int mi = 0; mi < 4; ++mi) {
                const _Float16* p =
                    &sA[cur][(waveM * 64 + mi * 16 + ln) * LDT + hf * 16];
                v8h lo = *(const v8h*)p;
                v8h hi = *(const v8h*)(p + 8);
                afrag[mi] = __builtin_shufflevector(lo, hi,
                    0, 1, 2, 3, 4, 5, 6, 7, 8, 9, 10, 11, 12, 13, 14, 15);
            }
            v16h bfrag[4];
#pragma unroll
            for (int ni = 0; ni < 4; ++ni) {
                const _Float16* p =
                    &sB[cur][(waveN * 64 + ni * 16 + ln) * LDT + hf * 16];
                v8h lo = *(const v8h*)p;
                v8h hi = *(const v8h*)(p + 8);
                bfrag[ni] = __builtin_shufflevector(lo, hi,
                    0, 1, 2, 3, 4, 5, 6, 7, 8, 9, 10, 11, 12, 13, 14, 15);
            }

#pragma unroll
            for (int mi = 0; mi < 4; ++mi)
#pragma unroll
                for (int ni = 0; ni < 4; ++ni)
                    acc[mi][ni] = __builtin_amdgcn_wmma_f32_16x16x32_f16(
                        false, afrag[mi], false, bfrag[ni],
                        (short)0, acc[mi][ni], false, false);
        }
    } else {
        // ---- fallback: synchronous staging with inline int8->fp16 dequant ----
        for (int kt = 0; kt < KTILES; ++kt) {
            const int k0 = kt * BK;
#pragma unroll
            for (int i = 0; i < 4; ++i) {
                const int row = srow + i * 32;
                v8h a = *(const v8h*)(x + (size_t)(mBase + row) * K_TOT + k0 + scol);
                *(v8h*)(&sA[0][row * LDT + scol]) = a;
            }
            const int* wg = wi + (size_t)(nBase + tid) * K_TOT + k0;
#pragma unroll
            for (int j = 0; j < 4; ++j) {
                v4i w0 = *(const v4i*)(wg + j * 8);
                v4i w1 = *(const v4i*)(wg + j * 8 + 4);
                v8h h;
#pragma unroll
                for (int e = 0; e < 4; ++e) {
                    h[e]     = (_Float16)w0[e];
                    h[4 + e] = (_Float16)w1[e];
                }
                *(v8h*)(&sB[0][tid * LDT + j * 8]) = h;
            }
            __syncthreads();

            v16h afrag[4];
#pragma unroll
            for (int mi = 0; mi < 4; ++mi) {
                const _Float16* p =
                    &sA[0][(waveM * 64 + mi * 16 + ln) * LDT + hf * 16];
                v8h lo = *(const v8h*)p;
                v8h hi = *(const v8h*)(p + 8);
                afrag[mi] = __builtin_shufflevector(lo, hi,
                    0, 1, 2, 3, 4, 5, 6, 7, 8, 9, 10, 11, 12, 13, 14, 15);
            }
            v16h bfrag[4];
#pragma unroll
            for (int ni = 0; ni < 4; ++ni) {
                const _Float16* p =
                    &sB[0][(waveN * 64 + ni * 16 + ln) * LDT + hf * 16];
                v8h lo = *(const v8h*)p;
                v8h hi = *(const v8h*)(p + 8);
                bfrag[ni] = __builtin_shufflevector(lo, hi,
                    0, 1, 2, 3, 4, 5, 6, 7, 8, 9, 10, 11, 12, 13, 14, 15);
            }
#pragma unroll
            for (int mi = 0; mi < 4; ++mi)
#pragma unroll
                for (int ni = 0; ni < 4; ++ni)
                    acc[mi][ni] = __builtin_amdgcn_wmma_f32_16x16x32_f16(
                        false, afrag[mi], false, bfrag[ni],
                        (short)0, acc[mi][ni], false, false);
            __syncthreads();
        }
    }

    // ---- epilogue: fused scale/bias in fp32, store fp16 ----
    // D layout: VGPR r, lane -> C[M = 8*hf + r][N = ln]
#pragma unroll
    for (int ni = 0; ni < 4; ++ni) {
        const int   n  = nBase + waveN * 64 + ni * 16 + ln;
        const float sc = (float)scales[n];
        const float bs = (float)bias[n];
#pragma unroll
        for (int mi = 0; mi < 4; ++mi) {
            const int m0 = mBase + waveM * 64 + mi * 16 + hf * 8;
#pragma unroll
            for (int r = 0; r < 8; ++r)
                out[(size_t)(m0 + r) * N_TOT + n] =
                    (_Float16)(acc[mi][ni][r] * sc + bs);
        }
    }
}

// ---------------------------------------------------------------------------
// Host-side launcher
// ---------------------------------------------------------------------------
extern "C" void kernel_launch(void* const* d_in, const int* in_sizes, int n_in,
                              void* d_out, int out_size, void* d_ws, size_t ws_size,
                              hipStream_t stream) {
    (void)in_sizes; (void)n_in; (void)out_size;

    const _Float16* x  = (const _Float16*)d_in[0];
    const int*      w  = (const int*)d_in[1];
    const _Float16* sc = (const _Float16*)d_in[2];
    const _Float16* bi = (const _Float16*)d_in[3];
    _Float16*       o  = (_Float16*)d_out;

    dim3 grid(NGRID_N * NGRID_M);        // 8192 blocks, swizzled in-kernel
    dim3 block(128);                     // 4 waves, 64x64 tile each

    const size_t wBytes = (size_t)N_TOT * K_TOT * sizeof(_Float16);  // 128 MB
    if (ws_size >= wBytes) {
        const int n4 = (N_TOT / 4) * K_TOT;  // 16M vec4 elements
        dequant_w8_f16<<<n4 / 256, 256, 0, stream>>>((const v4i*)w, (v4h*)d_ws, n4);
        w8a16_gemm<false><<<grid, block, 0, stream>>>(x, d_ws, sc, bi, o);
    } else {
        w8a16_gemm<true><<<grid, block, 0, stream>>>(x, w, sc, bi, o);
    }
}
